// ODERNN_encoder_44616120271297
// MI455X (gfx1250) — matrix operand order repaired and checked by reading, not compiled, not run on hardware
//
#include <hip/hip_runtime.h>
#include <hip/hip_bf16.h>

// ODE-RNN encoder for MI455X (gfx1250): bf16 WMMA GEMMs (f32 accum) for all
// matmuls; weights converted once to bf16 (L2-resident ~3.9MB); sequential
// scan realized as a per-step kernel chain on `stream`.
//
// GEMM: one wave = 32x32 C tile (4 accumulators). Software-pipelined K loop
// with the last chunk peeled: loads for chunk k+1 issue before the WMMAs of
// chunk k, with no conditional defs (avoids spills/zero-init in the loop).

typedef __attribute__((ext_vector_type(16))) __bf16 v16bf;
typedef __attribute__((ext_vector_type(8)))  __bf16 v8bf;
typedef __attribute__((ext_vector_type(8)))  float  v8f;

#define TT 128
#define BB 512
#define II 64
#define HH 512
#define H2 1024
#define H3 1536

__device__ __forceinline__ v16bf odernn_cat(v8bf lo, v8bf hi)
{
    return __builtin_shufflevector(lo, hi, 0, 1, 2, 3, 4, 5, 6, 7,
                                           8, 9, 10, 11, 12, 13, 14, 15);
}

// Fragment loaders per CDNA5 ISA 7.12.2 (bf16 16x16x32):
//   A (16x32): lane%16 = row; half=lane/16 selects K offset {0,8}; elements
//              0..7 = K base+0..7, elements 8..15 = K base+16..23.
//   B (32x16): lane%16 = column; same K order as A (half selects K offset {0,8};
//              elements 0..7 = K base+0..7, elements 8..15 = K base+16..23).
__device__ __forceinline__ v16bf odernn_load_afrag(const __bf16* __restrict__ p)
{
    return odernn_cat(*reinterpret_cast<const v8bf*>(p),
                      *reinterpret_cast<const v8bf*>(p + 16));
}
__device__ __forceinline__ v16bf odernn_load_bfrag(const __bf16* __restrict__ p)
{
    return odernn_cat(*reinterpret_cast<const v8bf*>(p),
                      *reinterpret_cast<const v8bf*>(p + 16));
}

__device__ __forceinline__ v8f odernn_wmma(v16bf a, v16bf b, v8f c)
{
    return __builtin_amdgcn_wmma_f32_16x16x32_bf16(
        /*neg_a=*/false, a, /*neg_b=*/false, b,
        /*c_mod=*/(short)0, c, /*reuse_a=*/false, /*reuse_b=*/false);
}

// ---------------------------------------------------------------------------
// WMMA GEMM:  C[M,N] = A[M,K](bf16,row-major) * W[N,K](bf16,row-major)^T + bias
// MODE 0: write f32 C.  MODE 1: write bf16(tanh(C)).
// ---------------------------------------------------------------------------
template<int MODE>
__global__ __launch_bounds__(256, 1)
void odernn_wmma_gemm_nt(const __bf16* __restrict__ A, int lda,
                         const __bf16* __restrict__ W, int ldb,
                         const float*  __restrict__ bias,
                         float*  __restrict__ outF,
                         __bf16* __restrict__ outB, int ldc,
                         int M, int N, int K)
{
    const int wave   = blockIdx.x * (blockDim.x >> 5) + (threadIdx.x >> 5);
    const int tilesM = M >> 5;                 // 32 rows per wave tile
    const int tilesN = N >> 5;                 // 32 cols per wave tile
    if (wave >= tilesM * tilesN) return;       // wave-uniform guard (EXEC all-1)

    const int mTile = wave % tilesM;
    const int nTile = wave / tilesM;
    const int lane  = threadIdx.x & 31;
    const int half  = lane >> 4;               // 0 or 1
    const int l16   = lane & 15;

    const __bf16* __restrict__ aP0 = A + (size_t)(mTile * 32 + l16)      * lda + half * 8;
    const __bf16* __restrict__ aP1 = A + (size_t)(mTile * 32 + 16 + l16) * lda + half * 8;
    const __bf16* __restrict__ bP0 = W + (size_t)(nTile * 32 + l16)      * ldb + half * 8;
    const __bf16* __restrict__ bP1 = W + (size_t)(nTile * 32 + 16 + l16) * ldb + half * 8;

    v8f c00 = {}, c01 = {}, c10 = {}, c11 = {};

    // Software pipeline: preload chunk 0; steady-state iterations load chunk
    // k+1 unconditionally before consuming chunk k; last chunk peeled.
    v16bf a0 = odernn_load_afrag(aP0);
    v16bf a1 = odernn_load_afrag(aP1);
    v16bf b0 = odernn_load_bfrag(bP0);
    v16bf b1 = odernn_load_bfrag(bP1);

#pragma unroll 2
    for (int k = 32; k < K; k += 32) {
        v16bf a0n = odernn_load_afrag(aP0 + k);
        v16bf a1n = odernn_load_afrag(aP1 + k);
        v16bf b0n = odernn_load_bfrag(bP0 + k);
        v16bf b1n = odernn_load_bfrag(bP1 + k);
        c00 = odernn_wmma(a0, b0, c00);
        c01 = odernn_wmma(a0, b1, c01);
        c10 = odernn_wmma(a1, b0, c10);
        c11 = odernn_wmma(a1, b1, c11);
        a0 = a0n; a1 = a1n; b0 = b0n; b1 = b1n;
    }
    c00 = odernn_wmma(a0, b0, c00);
    c01 = odernn_wmma(a0, b1, c01);
    c10 = odernn_wmma(a1, b0, c10);
    c11 = odernn_wmma(a1, b1, c11);

    // Epilogue. C layout (f32 16x16): lane%16 = column, VGPR v = row v+8*half.
    const int nBase = nTile * 32 + l16;
    const int mBase = mTile * 32 + half * 8;
    const float bv0 = bias ? bias[nBase]      : 0.0f;
    const float bv1 = bias ? bias[nBase + 16] : 0.0f;
#pragma unroll
    for (int v = 0; v < 8; ++v) {
        const int m0 = mBase + v;
        const int m1 = mBase + 16 + v;
        float v00 = c00[v] + bv0, v01 = c01[v] + bv1;
        float v10 = c10[v] + bv0, v11 = c11[v] + bv1;
        if (MODE == 0) {
            outF[(size_t)m0 * ldc + nBase]      = v00;
            outF[(size_t)m0 * ldc + nBase + 16] = v01;
            outF[(size_t)m1 * ldc + nBase]      = v10;
            outF[(size_t)m1 * ldc + nBase + 16] = v11;
        } else {
            outB[(size_t)m0 * ldc + nBase]      = (__bf16)tanhf(v00);
            outB[(size_t)m0 * ldc + nBase + 16] = (__bf16)tanhf(v01);
            outB[(size_t)m1 * ldc + nBase]      = (__bf16)tanhf(v10);
            outB[(size_t)m1 * ldc + nBase + 16] = (__bf16)tanhf(v11);
        }
    }
}

// ---------------------------------------------------------------------------
// Elementwise helpers
// ---------------------------------------------------------------------------
__global__ void odernn_f32_to_bf16(const float* __restrict__ src,
                                   __bf16* __restrict__ dst, int n)
{
    int i = blockIdx.x * blockDim.x + threadIdx.x;
    if (i < n) dst[i] = (__bf16)src[i];
}

__global__ void odernn_zero_h(float* __restrict__ h, __bf16* __restrict__ hbf, int n)
{
    int i = blockIdx.x * blockDim.x + threadIdx.x;
    if (i < n) { h[i] = 0.0f; hbf[i] = (__bf16)0.0f; }
}

// GRU gate fusion: h_rnn = (1-z)*n + z*h  -> states[t], and seed RK4 state.
__global__ void odernn_gru_gates(const float* __restrict__ gi,
                                 const float* __restrict__ gh,
                                 const float* __restrict__ h,
                                 float*  __restrict__ state_out,  // d_out + t*B*H
                                 float*  __restrict__ hr,
                                 float*  __restrict__ y,
                                 __bf16* __restrict__ ybf,
                                 float*  __restrict__ acc, int n)
{
    int i = blockIdx.x * blockDim.x + threadIdx.x;
    if (i >= n) return;
    int b = i / HH, j = i - b * HH;
    size_t base = (size_t)b * H3 + j;
    float ir = gi[base], iz = gi[base + HH], inn = gi[base + 2 * HH];
    float hrr = gh[base], hz = gh[base + HH], hn = gh[base + 2 * HH];
    float r  = 1.0f / (1.0f + __expf(-(ir + hrr)));
    float z  = 1.0f / (1.0f + __expf(-(iz + hz)));
    float nn = tanhf(inn + r * hn);
    float v  = (1.0f - z) * nn + z * h[i];
    state_out[i] = v;
    hr[i]  = v;
    y[i]   = v;
    ybf[i] = (__bf16)v;
    acc[i] = 0.0f;
}

// RK4 stage combine: orth-projection (row dot via LDS reduction), k = g-proj*y,
// acc += w*k; next y = hr + cnext*dt*k  (or final h = hr + dt/6 * acc).
__global__ void odernn_rk4_combine(const float* __restrict__ g,
                                   float*  __restrict__ y,      // in/out
                                   __bf16* __restrict__ ybf,
                                   const float* __restrict__ hr,
                                   float*  __restrict__ acc,
                                   float*  __restrict__ h,
                                   __bf16* __restrict__ hbf,
                                   const float* __restrict__ dt_ptr,
                                   float wgt, float cnext, int last)
{
    const int row = blockIdx.x;          // one block (256 thr = 8 waves) per row
    const int tid = threadIdx.x;
    const float* __restrict__ gr = g + (size_t)row * HH;
    float* __restrict__ yr = y + (size_t)row * HH;

    __shared__ float s1[256], s2[256];
    float gy = 0.0f, yy = 0.0f;
    for (int j = tid; j < HH; j += 256) {
        float yv = yr[j];
        gy += gr[j] * yv;
        yy += yv * yv;
    }
    s1[tid] = gy; s2[tid] = yy;
    __syncthreads();
    for (int off = 128; off > 0; off >>= 1) {
        if (tid < off) { s1[tid] += s1[tid + off]; s2[tid] += s2[tid + off]; }
        __syncthreads();
    }
    const float proj = s1[0] / (s2[0] + 1e-8f);
    const float dt   = *dt_ptr;

    for (int j = tid; j < HH; j += 256) {
        size_t idx = (size_t)row * HH + j;
        float yv = yr[j];
        float k  = gr[j] - proj * yv;
        float a  = acc[idx] + wgt * k;
        acc[idx] = a;
        if (!last) {
            float yn = hr[idx] + cnext * dt * k;
            yr[j]    = yn;
            ybf[idx] = (__bf16)yn;
        } else {
            float hn = hr[idx] + (dt * (1.0f / 6.0f)) * a;
            h[idx]   = hn;
            hbf[idx] = (__bf16)hn;
        }
    }
}

// ---------------------------------------------------------------------------
extern "C" void kernel_launch(void* const* d_in, const int* in_sizes, int n_in,
                              void* d_out, int out_size, void* d_ws, size_t ws_size,
                              hipStream_t stream)
{
    const float* seq  = (const float*)d_in[0];  // [T,B,I]
    const float* dts  = (const float*)d_in[1];  // [T]
    const float* w_ih = (const float*)d_in[2];  // [3H,I]
    const float* w_hh = (const float*)d_in[3];  // [3H,H]
    const float* b_ih = (const float*)d_in[4];  // [3H]
    const float* b_hh = (const float*)d_in[5];  // [3H]
    const float* w1   = (const float*)d_in[6];  // [2H,H]
    const float* b1   = (const float*)d_in[7];  // [2H]
    const float* w2   = (const float*)d_in[8];  // [H,2H]
    const float* b2   = (const float*)d_in[9];  // [H]
    float* out = (float*)d_out;                 // [T,B,H]

    // Workspace carve-out (256B aligned)
    char*  ws  = (char*)d_ws;
    size_t off = 0;
    auto carve = [&](size_t bytes) -> char* {
        char* p = ws + off;
        off = (off + bytes + 255) & ~(size_t)255;
        return p;
    };
    __bf16* wih_bf = (__bf16*)carve((size_t)H3 * II * 2);
    __bf16* whh_bf = (__bf16*)carve((size_t)H3 * HH * 2);
    __bf16* w1_bf  = (__bf16*)carve((size_t)H2 * HH * 2);
    __bf16* w2_bf  = (__bf16*)carve((size_t)HH * H2 * 2);
    __bf16* x_bf   = (__bf16*)carve((size_t)BB * II * 2);
    __bf16* h_bf   = (__bf16*)carve((size_t)BB * HH * 2);
    __bf16* y_bf   = (__bf16*)carve((size_t)BB * HH * 2);
    __bf16* u_bf   = (__bf16*)carve((size_t)BB * H2 * 2);
    float*  h      = (float*) carve((size_t)BB * HH * 4);
    float*  gi     = (float*) carve((size_t)BB * H3 * 4);
    float*  gh     = (float*) carve((size_t)BB * H3 * 4);
    float*  hr     = (float*) carve((size_t)BB * HH * 4);
    float*  y      = (float*) carve((size_t)BB * HH * 4);
    float*  gbuf   = (float*) carve((size_t)BB * HH * 4);
    float*  acc    = (float*) carve((size_t)BB * HH * 4);
    (void)ws_size; (void)in_sizes; (void)n_in; (void)out_size;

    const int BH = BB * HH;

    // One-time (per call) setup: bf16 weights + zero hidden state
    odernn_f32_to_bf16<<<(H3 * II + 255) / 256, 256, 0, stream>>>(w_ih, wih_bf, H3 * II);
    odernn_f32_to_bf16<<<(H3 * HH + 255) / 256, 256, 0, stream>>>(w_hh, whh_bf, H3 * HH);
    odernn_f32_to_bf16<<<(H2 * HH + 255) / 256, 256, 0, stream>>>(w1,   w1_bf,  H2 * HH);
    odernn_f32_to_bf16<<<(HH * H2 + 255) / 256, 256, 0, stream>>>(w2,   w2_bf,  HH * H2);
    odernn_zero_h<<<(BH + 255) / 256, 256, 0, stream>>>(h, h_bf, BH);

    const float rkw[4] = {1.0f, 2.0f, 2.0f, 1.0f};
    const float rkc[4] = {0.5f, 0.5f, 1.0f, 0.0f};

    // 32x32 tiles, 8 waves (256 threads) per block
    const int blocksGRU  = (16 * 48) / 8;  // [512 x 1536] -> 768 waves
    const int blocksMLP1 = (16 * 32) / 8;  // [512 x 1024] -> 512 waves
    const int blocksMLP2 = (16 * 16) / 8;  // [512 x  512] -> 256 waves

    for (int t = 0; t < TT; ++t) {
        // x_t -> bf16
        odernn_f32_to_bf16<<<(BB * II + 255) / 256, 256, 0, stream>>>(
            seq + (size_t)t * BB * II, x_bf, BB * II);

        // gi = x @ W_ih^T + b_ih   [512,1536] K=64
        odernn_wmma_gemm_nt<0><<<blocksGRU, 256, 0, stream>>>(
            x_bf, II, wih_bf, II, b_ih, gi, nullptr, H3, BB, H3, II);
        // gh = h @ W_hh^T + b_hh   [512,1536] K=512
        odernn_wmma_gemm_nt<0><<<blocksGRU, 256, 0, stream>>>(
            h_bf, HH, whh_bf, HH, b_hh, gh, nullptr, H3, BB, H3, HH);

        // GRU gates -> states[t] and RK4 seed
        odernn_gru_gates<<<(BH + 255) / 256, 256, 0, stream>>>(
            gi, gh, h, out + (size_t)t * BH, hr, y, y_bf, acc, BH);

        // RK4 stages
        for (int s = 0; s < 4; ++s) {
            // u = tanh(y @ W1^T + b1) -> bf16  [512,1024] K=512
            odernn_wmma_gemm_nt<1><<<blocksMLP1, 256, 0, stream>>>(
                y_bf, HH, w1_bf, HH, b1, nullptr, u_bf, H2, BB, H2, HH);
            // g = u @ W2^T + b2  [512,512] K=1024
            odernn_wmma_gemm_nt<0><<<blocksMLP2, 256, 0, stream>>>(
                u_bf, H2, w2_bf, H2, b2, gbuf, nullptr, HH, BB, HH, H2);
            // projection + stage update (one block per batch row)
            odernn_rk4_combine<<<BB, 256, 0, stream>>>(
                gbuf, y, y_bf, hr, acc, h, h_bf, dts + t,
                rkw[s], rkc[s], (s == 3) ? 1 : 0);
        }
    }
}
